// SparseAttentionModel_7834020348442
// MI455X (gfx1250) — matrix operand order repaired and checked by reading, not compiled, not run on hardware
//
#include <hip/hip_runtime.h>
#include <hip/hip_bf16.h>

#define BATCH 2
#define SEQ 2048
#define DIM 1024
#define VOCAB 32000
#define CHUNK 256
#define NCHUNK (SEQ / CHUNK)      // 8
#define LCTX 4
#define LOCAL_WIN 512
#define GSTRIDE 1024
#define KLEN_PAD 896              // >= max klen (770); multiple of 128 (N tile) and 32 (K step)
#define NZ (BATCH * NCHUNK)       // 16 (z = b*8 + chunk)
#define MROWS (BATCH * SEQ)       // 4096

typedef __bf16 bf16;
typedef __bf16 v16bf __attribute__((ext_vector_type(16)));
typedef __bf16 v8bf  __attribute__((ext_vector_type(8)));
typedef __bf16 v4bf  __attribute__((ext_vector_type(4)));
typedef float  v8f   __attribute__((ext_vector_type(8)));

// ---------------------------------------------------------------------------
// Compile-time replica of the reference's deterministic sparse cache pruning.
// ---------------------------------------------------------------------------
struct KposTable {
  int pos[NCHUNK][KLEN_PAD];
  int len[NCHUNK];
};

constexpr KposTable make_kpos() {
  KposTable t{};
  int cache[SEQ] = {};
  int clen = 0;
  for (int i = 0; i < NCHUNK; ++i) {
    int full[SEQ] = {};
    int flen = 0;
    for (int j = 0; j < clen; ++j) full[flen++] = cache[j];
    for (int p = i * CHUNK; p < (i + 1) * CHUNK; ++p) full[flen++] = p;
    t.len[i] = flen;
    for (int j = 0; j < KLEN_PAD; ++j) t.pos[i][j] = (j < flen) ? full[j] : -1;
    int total_seen = (i + 1) * CHUNK;
    int local_start = total_seen - LOCAL_WIN;
    if (local_start < 0) local_start = 0;
    clen = 0;
    for (int j = 0; j < flen; ++j)
      if (full[j] >= local_start || (full[j] % GSTRIDE == 0)) cache[clen++] = full[j];
  }
  return t;
}

constexpr KposTable KPOS = make_kpos();

// ---------------------------------------------------------------------------
// Embedding + local-context sum  ->  ctx (bf16, MROWS x DIM)
// ---------------------------------------------------------------------------
__global__ __launch_bounds__(256)
void ctx_embed_kernel(const int* __restrict__ x, const float* __restrict__ emb,
                      bf16* __restrict__ ctx) {
  const int row = blockIdx.x;             // 0..MROWS-1
  const int b = row / SEQ, s = row % SEQ;
  const int d = threadIdx.x * 4;
  float4 acc = make_float4(0.f, 0.f, 0.f, 0.f);
  for (int off = 0; off < LCTX; ++off) {
    const int sp = s - off;
    if (sp < 0) break;
    const int tok = x[b * SEQ + sp];
    const float4 e = *(const float4*)(emb + (size_t)tok * DIM + d);
    acc.x += e.x; acc.y += e.y; acc.z += e.z; acc.w += e.w;
  }
  v4bf o;
  o[0] = (bf16)acc.x; o[1] = (bf16)acc.y; o[2] = (bf16)acc.z; o[3] = (bf16)acc.w;
  *(v4bf*)(ctx + (size_t)row * DIM + d) = o;
}

// ---------------------------------------------------------------------------
// f32 -> bf16 elementwise convert (weights)
// ---------------------------------------------------------------------------
__global__ __launch_bounds__(256)
void f32_to_bf16_kernel(const float* __restrict__ src, bf16* __restrict__ dst,
                        long long n) {
  const long long i = (long long)blockIdx.x * blockDim.x + threadIdx.x;
  if (i < n) dst[i] = (bf16)src[i];
}

// ---------------------------------------------------------------------------
// Generic batched bf16 WMMA GEMM:  D[z] = A[z](MxK) * B[z](NxK)^T + bias
// 128 threads = 4 waves; block tile 64(M) x 128(N); wave tile 32x64
// (2 m-frags x 4 n-frags = 8 WMMAs per 32-wide k-step). NO bounds guards:
// all callers guarantee M%64==0, N%128==0, K%32==0.
// A layout: lane L -> row M=L%16, contiguous 16-K block at (L/16)*16.
// B layout, same k order as A: lane L -> row N=L%16, contiguous 16-K block at (L/16)*16.
// D layout per ISA: v8f elem r -> (m = r + 8*(L/16), n = L%16).
// ---------------------------------------------------------------------------
#define MT 2
#define NT 4

__global__ __launch_bounds__(128)
void gemm_bf16_wmma(const bf16* __restrict__ A, long long sAz, int lda,
                    const bf16* __restrict__ B, long long sBz, int ldb,
                    float* __restrict__ Cf, bf16* __restrict__ Cb,
                    long long sCz, int ldc,
                    const float* __restrict__ bias,
                    int M, int N, int K) {
  const int lane = threadIdx.x & 31;
  const int wave = threadIdx.x >> 5;
  const int z = blockIdx.z;
  const int m0 = blockIdx.y * 64 + (wave >> 1) * 32;
  const int n0 = blockIdx.x * 128 + (wave & 1) * 64;
  const int lrow = lane & 15;
  const int lgrp = lane >> 4;

  const bf16* Az = A + (size_t)z * sAz;
  const bf16* Bz = B + (size_t)z * sBz;

  // Per-lane base pointers (K advances by 32 each step).
  const bf16* apA[MT];
#pragma unroll
  for (int mt = 0; mt < MT; ++mt)
    apA[mt] = Az + (size_t)(m0 + mt * 16 + lrow) * lda + lgrp * 16;
  const bf16* bpB[NT];
#pragma unroll
  for (int nt = 0; nt < NT; ++nt)
    bpB[nt] = Bz + (size_t)(n0 + nt * 16 + lrow) * ldb + lgrp * 16;

  v8f acc[MT][NT];
#pragma unroll
  for (int i = 0; i < MT; ++i)
#pragma unroll
    for (int j = 0; j < NT; ++j) acc[i][j] = {};

  for (int kk = 0; kk < K; kk += 32) {
    v16bf af[MT], bfg[NT];
#pragma unroll
    for (int mt = 0; mt < MT; ++mt) {
      const v8bf lo = *(const v8bf*)(apA[mt] + kk);
      const v8bf hi = *(const v8bf*)(apA[mt] + kk + 8);
#pragma unroll
      for (int t = 0; t < 8; ++t) { af[mt][t] = lo[t]; af[mt][8 + t] = hi[t]; }
    }
#pragma unroll
    for (int nt = 0; nt < NT; ++nt)
      bfg[nt] = *(const v16bf*)(bpB[nt] + kk);
#pragma unroll
    for (int mt = 0; mt < MT; ++mt)
#pragma unroll
      for (int nt = 0; nt < NT; ++nt)
        acc[mt][nt] = __builtin_amdgcn_wmma_f32_16x16x32_bf16(
            false, af[mt], false, bfg[nt], (short)0, acc[mt][nt], false, false);
  }

  const int mrow_off = 8 * lgrp;
#pragma unroll
  for (int mt = 0; mt < MT; ++mt) {
#pragma unroll
    for (int nt = 0; nt < NT; ++nt) {
      const int ncol = n0 + nt * 16 + lrow;
      const float bv = bias ? bias[ncol] : 0.0f;
#pragma unroll
      for (int r = 0; r < 8; ++r) {
        const int m = m0 + mt * 16 + r + mrow_off;
        const float v = acc[mt][nt][r] + bv;
        const size_t off = (size_t)z * sCz + (size_t)m * ldc + ncol;
        if (Cf) Cf[off] = v;
        else    Cb[off] = (bf16)v;
      }
    }
  }
}

// ---------------------------------------------------------------------------
// Gather K rows (row-major klen x DIM, zero-padded) and V^T (DIM x KLEN_PAD,
// zero-padded) per z = b*8 + chunk.
// ---------------------------------------------------------------------------
__global__ __launch_bounds__(256)
void gather_kv_kernel(const bf16* __restrict__ Kb, const bf16* __restrict__ Vb,
                      bf16* __restrict__ kg, bf16* __restrict__ vT) {
  const int zz = blockIdx.z;              // z = b*8 + ic
  const int ic = zz % NCHUNK, b = zz / NCHUNK;
  const int j = blockIdx.x;               // 0..KLEN_PAD-1
  const int d = threadIdx.x * 4;
  bf16* kgr = kg + ((size_t)zz * KLEN_PAD + j) * DIM + d;
  bf16* vTr = vT + (size_t)zz * DIM * KLEN_PAD;
  if (j < KPOS.len[ic]) {
    const int kp = KPOS.pos[ic][j];
    const size_t src = (size_t)(b * SEQ + kp) * DIM + d;
    *(v4bf*)kgr = *(const v4bf*)(Kb + src);
    const v4bf vv = *(const v4bf*)(Vb + src);
#pragma unroll
    for (int t = 0; t < 4; ++t) vTr[(size_t)(d + t) * KLEN_PAD + j] = vv[t];
  } else {
    v4bf z4;
#pragma unroll
    for (int t = 0; t < 4; ++t) z4[t] = (bf16)0.0f;
    *(v4bf*)kgr = z4;
#pragma unroll
    for (int t = 0; t < 4; ++t) vTr[(size_t)(d + t) * KLEN_PAD + j] = (bf16)0.0f;
  }
}

// ---------------------------------------------------------------------------
// Masked softmax over one row of scores (one wave32 per row).
// Row index: row = z*CHUNK + q, z = b*8 + ic.
// ---------------------------------------------------------------------------
__global__ __launch_bounds__(256)
void softmax_kernel(const float* __restrict__ scores, bf16* __restrict__ probs) {
  const int lane = threadIdx.x & 31;
  const int wid = threadIdx.x >> 5;
  const int row = blockIdx.x * 8 + wid;   // 0..NZ*CHUNK-1
  const int q = row % CHUNK;
  const int zz = row / CHUNK;
  const int ic = zz % NCHUNK;
  const int qpos = ic * CHUNK + q;
  const int klen = KPOS.len[ic];
  const float scale = 0.03125f;           // 1/sqrt(1024)
  const float* srow = scores + (size_t)row * KLEN_PAD;

  float ev[KLEN_PAD / 32];
  float mx = -3.0e38f;
#pragma unroll
  for (int t = 0; t < KLEN_PAD / 32; ++t) {
    const int j = t * 32 + lane;
    const bool valid = (j < klen) && (KPOS.pos[ic][j] <= qpos);
    const float s = valid ? srow[j] * scale : -3.0e38f;
    ev[t] = s;
    mx = fmaxf(mx, s);
  }
#pragma unroll
  for (int o = 16; o > 0; o >>= 1) mx = fmaxf(mx, __shfl_xor(mx, o, 32));
  float sum = 0.f;
#pragma unroll
  for (int t = 0; t < KLEN_PAD / 32; ++t) {
    const float e = (ev[t] > -1.0e38f) ? __expf(ev[t] - mx) : 0.f;
    ev[t] = e;
    sum += e;
  }
#pragma unroll
  for (int o = 16; o > 0; o >>= 1) sum += __shfl_xor(sum, o, 32);
  const float inv = 1.0f / sum;
  bf16* prow = probs + (size_t)row * KLEN_PAD;
#pragma unroll
  for (int t = 0; t < KLEN_PAD / 32; ++t)
    prow[t * 32 + lane] = (bf16)(ev[t] * inv);
}

// ---------------------------------------------------------------------------
// LayerNorm over DIM, write bf16 (one 256-thread block per row, 4 elems/thr).
// ---------------------------------------------------------------------------
__global__ __launch_bounds__(256)
void layernorm_kernel(const float* __restrict__ h, const float* __restrict__ w,
                      const float* __restrict__ bb, bf16* __restrict__ out) {
  __shared__ float red[256];
  const int row = blockIdx.x;
  const int d0 = threadIdx.x * 4;
  const float4 v = *(const float4*)(h + (size_t)row * DIM + d0);
  red[threadIdx.x] = v.x + v.y + v.z + v.w;
  __syncthreads();
  for (int o = 128; o > 0; o >>= 1) {
    if (threadIdx.x < o) red[threadIdx.x] += red[threadIdx.x + o];
    __syncthreads();
  }
  const float mu = red[0] * (1.0f / DIM);
  __syncthreads();
  const float dx = v.x - mu, dy = v.y - mu, dz = v.z - mu, dw = v.w - mu;
  red[threadIdx.x] = dx * dx + dy * dy + dz * dz + dw * dw;
  __syncthreads();
  for (int o = 128; o > 0; o >>= 1) {
    if (threadIdx.x < o) red[threadIdx.x] += red[threadIdx.x + o];
    __syncthreads();
  }
  const float inv = rsqrtf(red[0] * (1.0f / DIM) + 1e-5f);
  const float4 wv = *(const float4*)(w + d0);
  const float4 bv = *(const float4*)(bb + d0);
  v4bf o4;
  o4[0] = (bf16)(dx * inv * wv.x + bv.x);
  o4[1] = (bf16)(dy * inv * wv.y + bv.y);
  o4[2] = (bf16)(dz * inv * wv.z + bv.z);
  o4[3] = (bf16)(dw * inv * wv.w + bv.w);
  *(v4bf*)(out + (size_t)row * DIM + d0) = o4;
}

// ---------------------------------------------------------------------------
// Launch
// ---------------------------------------------------------------------------
extern "C" void kernel_launch(void* const* d_in, const int* in_sizes, int n_in,
                              void* d_out, int out_size, void* d_ws, size_t ws_size,
                              hipStream_t stream) {
  (void)in_sizes; (void)n_in; (void)out_size; (void)ws_size;
  const int*   x      = (const int*)d_in[0];
  const float* emb    = (const float*)d_in[1];
  const float* q_w    = (const float*)d_in[2];
  const float* q_b    = (const float*)d_in[3];
  const float* k_w    = (const float*)d_in[4];
  const float* k_b    = (const float*)d_in[5];
  const float* v_w    = (const float*)d_in[6];
  const float* v_b    = (const float*)d_in[7];
  const float* o_w    = (const float*)d_in[8];
  const float* o_b    = (const float*)d_in[9];
  const float* ln_w   = (const float*)d_in[10];
  const float* ln_b   = (const float*)d_in[11];
  const float* out_w  = (const float*)d_in[12];
  const float* out_b  = (const float*)d_in[13];
  float* out = (float*)d_out;

  char* p = (char*)d_ws;
  auto alloc = [&](size_t bytes) -> void* {
    void* r = (void*)p;
    p += (bytes + 255) & ~(size_t)255;
    return r;
  };
  bf16*  ctx_b  = (bf16*)alloc((size_t)MROWS * DIM * 2);
  bf16*  qw_b   = (bf16*)alloc((size_t)DIM * DIM * 2);
  bf16*  kw_b   = (bf16*)alloc((size_t)DIM * DIM * 2);
  bf16*  vw_b   = (bf16*)alloc((size_t)DIM * DIM * 2);
  bf16*  ow_b   = (bf16*)alloc((size_t)DIM * DIM * 2);
  bf16*  outw_b = (bf16*)alloc((size_t)VOCAB * DIM * 2);
  bf16*  Qb     = (bf16*)alloc((size_t)MROWS * DIM * 2);
  bf16*  Kb     = (bf16*)alloc((size_t)MROWS * DIM * 2);
  bf16*  Vb     = (bf16*)alloc((size_t)MROWS * DIM * 2);
  bf16*  kg     = (bf16*)alloc((size_t)NZ * KLEN_PAD * DIM * 2);
  bf16*  vT     = (bf16*)alloc((size_t)NZ * DIM * KLEN_PAD * 2);
  float* sc     = (float*)alloc((size_t)NZ * CHUNK * KLEN_PAD * 4);
  bf16*  probs  = (bf16*)alloc((size_t)NZ * CHUNK * KLEN_PAD * 2);
  bf16*  att_b  = (bf16*)alloc((size_t)MROWS * DIM * 2);
  float* h_f32  = (float*)alloc((size_t)MROWS * DIM * 4);
  bf16*  hln_b  = (bf16*)alloc((size_t)MROWS * DIM * 2);

  // 1. embedding + local context sum (bf16)
  ctx_embed_kernel<<<MROWS, 256, 0, stream>>>(x, emb, ctx_b);

  // 2. weight conversion to bf16
  auto conv = [&](const float* s, bf16* d, long long n) {
    f32_to_bf16_kernel<<<(unsigned)((n + 255) / 256), 256, 0, stream>>>(s, d, n);
  };
  conv(q_w, qw_b, (long long)DIM * DIM);
  conv(k_w, kw_b, (long long)DIM * DIM);
  conv(v_w, vw_b, (long long)DIM * DIM);
  conv(o_w, ow_b, (long long)DIM * DIM);
  conv(out_w, outw_b, (long long)VOCAB * DIM);

  const dim3 blk(128);
  // 3. Q/K/V projections: [MROWS x DIM] = ctx * W^T + b  (bf16 out)
  gemm_bf16_wmma<<<dim3(DIM / 128, MROWS / 64, 1), blk, 0, stream>>>(
      ctx_b, 0LL, DIM, qw_b, 0LL, DIM, nullptr, Qb, 0LL, DIM, q_b, MROWS, DIM, DIM);
  gemm_bf16_wmma<<<dim3(DIM / 128, MROWS / 64, 1), blk, 0, stream>>>(
      ctx_b, 0LL, DIM, kw_b, 0LL, DIM, nullptr, Kb, 0LL, DIM, k_b, MROWS, DIM, DIM);
  gemm_bf16_wmma<<<dim3(DIM / 128, MROWS / 64, 1), blk, 0, stream>>>(
      ctx_b, 0LL, DIM, vw_b, 0LL, DIM, nullptr, Vb, 0LL, DIM, v_b, MROWS, DIM, DIM);

  // 4. gather sparse K rows and V^T per (batch, chunk)
  gather_kv_kernel<<<dim3(KLEN_PAD, 1, NZ), 256, 0, stream>>>(Kb, Vb, kg, vT);

  // 5. scores[z] = Q_chunk * Kg^T   (M=256, N=KLEN_PAD, K=DIM)
  gemm_bf16_wmma<<<dim3(KLEN_PAD / 128, CHUNK / 64, NZ), blk, 0, stream>>>(
      Qb, (long long)CHUNK * DIM, DIM,
      kg, (long long)KLEN_PAD * DIM, DIM,
      sc, nullptr, (long long)CHUNK * KLEN_PAD, KLEN_PAD,
      nullptr, CHUNK, KLEN_PAD, DIM);

  // 6. masked softmax -> probs (bf16, zero-padded)
  softmax_kernel<<<(NZ * CHUNK) / 8, 256, 0, stream>>>(sc, probs);

  // 7. attended[z] = probs * V     (M=256, N=DIM, K=KLEN_PAD), bf16 out
  gemm_bf16_wmma<<<dim3(DIM / 128, CHUNK / 64, NZ), blk, 0, stream>>>(
      probs, (long long)CHUNK * KLEN_PAD, KLEN_PAD,
      vT, (long long)DIM * KLEN_PAD, KLEN_PAD,
      nullptr, att_b, (long long)CHUNK * DIM, DIM,
      nullptr, CHUNK, DIM, KLEN_PAD);

  // 8. output projection: h = attended * o_w^T + o_b (f32 out for LN)
  gemm_bf16_wmma<<<dim3(DIM / 128, MROWS / 64, 1), blk, 0, stream>>>(
      att_b, 0LL, DIM, ow_b, 0LL, DIM, h_f32, nullptr, 0LL, DIM, o_b, MROWS, DIM, DIM);

  // 9. LayerNorm -> bf16
  layernorm_kernel<<<MROWS, 256, 0, stream>>>(h_f32, ln_w, ln_b, hln_b);

  // 10. vocab projection: out = h_ln * out_w^T + out_b (dominant GEMM, f32 out)
  gemm_bf16_wmma<<<dim3(VOCAB / 128, MROWS / 64, 1), blk, 0, stream>>>(
      hln_b, 0LL, DIM, outw_b, 0LL, DIM, out, nullptr, 0LL, VOCAB, out_b,
      MROWS, VOCAB, DIM);
}
